// PTv3_deteccion_76467597738041
// MI455X (gfx1250) — compile-verified, matrix operand order checked by reading, not run on hardware
//
#include <hip/hip_runtime.h>
#include <hip/hip_bf16.h>
#include <cstdint>

typedef __attribute__((ext_vector_type(16))) _Float16 v16h;
typedef __attribute__((ext_vector_type(8)))  float    v8f;

#define B_    4
#define NPTS  65536
#define BN    (B_ * NPTS)
#define PATCH 64

// ---------------------------------------------------------------------------
// helpers
// ---------------------------------------------------------------------------
__device__ __forceinline__ float gelu_tanh(float v) {
    float v3 = v * v * v;
    return 0.5f * v * (1.f + tanhf(0.7978845608028654f * (v + 0.044715f * v3)));
}

__device__ __forceinline__ int kmap16(int e, int kh) {
    // ISA 7.12.2: 16-bit A fragment, lanes 0-15 hold K={0..7,16..23}, lanes 16-31 K={8..15,24..31}
    return (e < 8) ? (kh * 8 + e) : (16 + kh * 8 + (e - 8));
}

// A fragment 16x32 (f16) from LDS f16, row-major, zero-padded beyond klim
__device__ __forceinline__ v16h frag_a_h(const _Float16* A, int lda, int m0, int k0,
                                         int klim, int lane) {
    v16h a;
    int row = m0 + (lane & 15);
    int kh  = lane >> 4;
#pragma unroll
    for (int e = 0; e < 16; ++e) {
        int k = k0 + kmap16(e, kh);
        a[e] = (k < klim) ? A[row * lda + k] : (_Float16)0.f;
    }
    return a;
}

// A fragment from LDS f32 (attention probabilities), converted to f16
__device__ __forceinline__ v16h frag_a_f(const float* A, int lda, int m0, int k0,
                                         int klim, int lane) {
    v16h a;
    int row = m0 + (lane & 15);
    int kh  = lane >> 4;
#pragma unroll
    for (int e = 0; e < 16; ++e) {
        int k = k0 + kmap16(e, kh);
        a[e] = (k < klim) ? (_Float16)A[row * lda + k] : (_Float16)0.f;
    }
    return a;
}

// B fragment 32x16 from pre-converted global f16 weights, row-major KxN
__device__ __forceinline__ v16h frag_b_gh(const _Float16* W, int ldn, int k0, int n0,
                                          int klim, int lane) {
    v16h b;
    int col = n0 + (lane & 15);
    int kh  = lane >> 4;
#pragma unroll
    for (int e = 0; e < 16; ++e) {
        int k = k0 + kmap16(e, kh);
        b[e] = (k < klim) ? W[k * ldn + col] : (_Float16)0.f;
    }
    return b;
}

// B fragment 32x16 from LDS f16, row-major KxN, with column limit (head dim pad)
__device__ __forceinline__ v16h frag_b_h(const _Float16* Bm, int ldb, int k0, int n0,
                                         int klim, int nlim, int lane) {
    v16h b;
    int col  = n0 + (lane & 15);
    int kh   = lane >> 4;
    bool okc = col < nlim;
#pragma unroll
    for (int e = 0; e < 16; ++e) {
        int k = k0 + kmap16(e, kh);
        b[e] = (okc && k < klim) ? Bm[k * ldb + col] : (_Float16)0.f;
    }
    return b;
}

__device__ __forceinline__ v8f wmma16(v16h a, v16h b, v8f c) {
    return __builtin_amdgcn_wmma_f32_16x16x32_f16(false, a, false, b, (short)0, c,
                                                  false, false);
}

// ---------------------------------------------------------------------------
// weight pre-conversion (one pass; weights then stay f16 + L2-resident)
// ---------------------------------------------------------------------------
__global__ void f32_to_f16_kernel(const float* __restrict__ src,
                                  _Float16* __restrict__ dst, int n) {
    int i = blockIdx.x * 256 + threadIdx.x;
    if (i < n) dst[i] = (_Float16)src[i];
}

// ---------------------------------------------------------------------------
// Morton sort
// ---------------------------------------------------------------------------
__device__ __forceinline__ unsigned part1by2(unsigned n) {
    n &= 1023u;
    n = (n | (n << 16)) & 50331903u;
    n = (n | (n << 8))  & 50393103u;
    n = (n | (n << 4))  & 51130563u;
    n = (n | (n << 2))  & 153391689u;
    return n;
}

__global__ void morton_init(const float* __restrict__ pts,
                            unsigned long long* __restrict__ keys) {
    int i = blockIdx.x * 256 + threadIdx.x;           // < BN
    const float* p = pts + (long long)i * 4;
    int gx = (int)floorf(p[0]); gx = gx < 0 ? 0 : (gx > 1023 ? 1023 : gx);
    int gy = (int)floorf(p[1]); gy = gy < 0 ? 0 : (gy > 1023 ? 1023 : gy);
    int gz = (int)floorf(p[2]); gz = gz < 0 ? 0 : (gz > 1023 ? 1023 : gz);
    unsigned m = part1by2((unsigned)gx) | (part1by2((unsigned)gy) << 1) |
                 (part1by2((unsigned)gz) << 2);
    keys[i] = ((unsigned long long)m << 16) | (unsigned)(i & (NPTS - 1));
}

__global__ void bitonic_step(unsigned long long* __restrict__ keys, int j, int k) {
    int i = blockIdx.x * 256 + threadIdx.x;           // < BN
    int b = i >> 16;
    int t = i & (NPTS - 1);
    int ixj = t ^ j;
    if (ixj > t) {
        unsigned long long* kb = keys + ((long long)b << 16);
        unsigned long long a = kb[t], c = kb[ixj];
        bool up = ((t & k) == 0);
        if ((a > c) == up) { kb[t] = c; kb[ixj] = a; }
    }
}

__global__ void embed_gather(const float* __restrict__ pts,
                             const unsigned long long* __restrict__ keys,
                             const float* __restrict__ W, const float* __restrict__ bias,
                             float* __restrict__ x0) {
    long long idx = (long long)blockIdx.x * 256 + threadIdx.x;   // < BN*32
    int c = (int)(idx & 31);
    long long row = idx >> 5;
    long long b = row >> 16;
    int src = (int)(keys[row] & 0xFFFFull);
    float f = pts[((b << 16) + src) * 4 + 3];
    x0[row * 32 + c] = f * W[c] + bias[c];
}

// ---------------------------------------------------------------------------
// Transformer block: one 64-token patch per 256-thread workgroup, 8 waves.
// All GEMMs run on v_wmma_f32_16x16x32_f16 (f16 in, f32 accumulate).
// Weights arrive pre-converted to f16 (half the operand bytes, no cvt in loop).
// ---------------------------------------------------------------------------
template <int C, int H>
__global__ void __launch_bounds__(256)
block_kernel(float* __restrict__ x,
             const float* __restrict__ ln1g, const float* __restrict__ ln1b,
             const _Float16* __restrict__ Wqkv,
             const _Float16* __restrict__ WoW,  const float* __restrict__ Wob,
             const float* __restrict__ ln2g, const float* __restrict__ ln2b,
             const _Float16* __restrict__ fc1W, const float* __restrict__ fc1b,
             const _Float16* __restrict__ fc2W, const float* __restrict__ fc2b) {
    constexpr int D   = C / H;               // head dim: 16/24/32 -> K-padded to 32
    constexpr int NTD = (D + 15) / 16;
    extern __shared__ char smem[];
    float*    xs  = (float*)smem;                               // 64*C   f32 (residual)
    _Float16* hs  = (_Float16*)(xs + PATCH * C);                // 64*C   f16 (ln out / o / ln2 out)
    _Float16* qs  = (_Float16*)(hs + PATCH * C);                // 64*3C  f16 (qkv / gelu out)
    float*    att = (float*)(qs + PATCH * 3 * C);               // H*64*64 f32

    const int tid  = threadIdx.x;
    const int lane = tid & 31;
    const int wv   = tid >> 5;
    const long long base = (long long)blockIdx.x * PATCH * C;

    if (tid == 0) {   // stage weights through L2 (global_prefetch_b8)
        __builtin_prefetch(Wqkv, 0, 3);
        __builtin_prefetch(WoW,  0, 3);
        __builtin_prefetch(fc1W, 0, 3);
        __builtin_prefetch(fc2W, 0, 3);
    }

    for (int i = tid; i < PATCH * C; i += 256) xs[i] = x[base + i];
    __syncthreads();

    // ---- LN1 -> hs (f16)
    if (tid < PATCH) {
        const float* r = xs + tid * C;
        float mu = 0.f;
        for (int c = 0; c < C; ++c) mu += r[c];
        mu /= C;
        float var = 0.f;
        for (int c = 0; c < C; ++c) { float d = r[c] - mu; var += d * d; }
        float rs = rsqrtf(var / C + 1e-5f);
        _Float16* o = hs + tid * C;
        for (int c = 0; c < C; ++c)
            o[c] = (_Float16)((r[c] - mu) * rs * ln1g[c] + ln1b[c]);
    }
    __syncthreads();

    // ---- QKV: (64xC) @ (Cx3C) -> qs (f16)
    {
        constexpr int NT = 3 * C / 16;
        for (int t = wv; t < 4 * NT; t += 8) {
            int mt = t & 3, nt = t >> 2;
            v8f acc = {};
            for (int kk = 0; kk < C; kk += 32) {
                v16h a = frag_a_h(hs, C, mt * 16, kk, C, lane);
                v16h b = frag_b_gh(Wqkv, 3 * C, kk, nt * 16, C, lane);
                acc = wmma16(a, b, acc);
            }
            int col = nt * 16 + (lane & 15);
#pragma unroll
            for (int r = 0; r < 8; ++r) {
                int row = mt * 16 + r + 8 * (lane >> 4);
                qs[row * 3 * C + col] = (_Float16)acc[r];
            }
        }
    }
    __syncthreads();

    // ---- scores: att[h] = (q @ k^T) / sqrt(D); K zero-padded to 32
    {
        const float scale = rsqrtf((float)D);
        for (int t = wv; t < H * 16; t += 8) {
            int h = t >> 4, tt = t & 15, mt = tt & 3, nt = tt >> 2;
            v16h a = frag_a_h(qs + h * D,     3 * C, mt * 16, 0, D, lane);
            v16h b = frag_a_h(qs + C + h * D, 3 * C, nt * 16, 0, D, lane); // k^T
            v8f acc = {};
            acc = wmma16(a, b, acc);
            float* ah = att + h * PATCH * PATCH;
            int col = nt * 16 + (lane & 15);
#pragma unroll
            for (int r = 0; r < 8; ++r) {
                int row = mt * 16 + r + 8 * (lane >> 4);
                ah[row * PATCH + col] = acc[r] * scale;
            }
        }
    }
    __syncthreads();

    // ---- softmax over each of the H*64 rows
    if (tid < H * PATCH) {
        float* rp = att + tid * PATCH;
        float mx = rp[0];
        for (int q = 1; q < PATCH; ++q) mx = fmaxf(mx, rp[q]);
        float s = 0.f;
        for (int q = 0; q < PATCH; ++q) { float e = expf(rp[q] - mx); rp[q] = e; s += e; }
        float inv = 1.f / s;
        for (int q = 0; q < PATCH; ++q) rp[q] *= inv;
    }
    __syncthreads();

    // ---- o = att @ v -> hs (f16), K = 64
    for (int t = wv; t < H * 4 * NTD; t += 8) {
        int h  = t / (4 * NTD);
        int tt = t % (4 * NTD);
        int mt = tt % 4, nt = tt / 4;
        const float* ah = att + h * PATCH * PATCH;
        v8f acc = {};
        for (int kk = 0; kk < PATCH; kk += 32) {
            v16h a = frag_a_f(ah, PATCH, mt * 16, kk, PATCH, lane);
            v16h b = frag_b_h(qs + 2 * C + h * D, 3 * C, kk, nt * 16, PATCH, D, lane);
            acc = wmma16(a, b, acc);
        }
        int col = nt * 16 + (lane & 15);
        if (col < D) {
#pragma unroll
            for (int r = 0; r < 8; ++r) {
                int row = mt * 16 + r + 8 * (lane >> 4);
                hs[row * C + h * D + col] = (_Float16)acc[r];
            }
        }
    }
    __syncthreads();

    // ---- x += o @ Wo + b
    for (int t = wv; t < 4 * (C / 16); t += 8) {
        int mt = t & 3, nt = t >> 2;
        v8f acc = {};
        for (int kk = 0; kk < C; kk += 32) {
            v16h a = frag_a_h(hs, C, mt * 16, kk, C, lane);
            v16h b = frag_b_gh(WoW, C, kk, nt * 16, C, lane);
            acc = wmma16(a, b, acc);
        }
        int col = nt * 16 + (lane & 15);
        float bia = Wob[col];
#pragma unroll
        for (int r = 0; r < 8; ++r) {
            int row = mt * 16 + r + 8 * (lane >> 4);
            xs[row * C + col] += acc[r] + bia;
        }
    }
    __syncthreads();

    // ---- LN2 -> hs (f16)
    if (tid < PATCH) {
        const float* r = xs + tid * C;
        float mu = 0.f;
        for (int c = 0; c < C; ++c) mu += r[c];
        mu /= C;
        float var = 0.f;
        for (int c = 0; c < C; ++c) { float d = r[c] - mu; var += d * d; }
        float rs = rsqrtf(var / C + 1e-5f);
        _Float16* o = hs + tid * C;
        for (int c = 0; c < C; ++c)
            o[c] = (_Float16)((r[c] - mu) * rs * ln2g[c] + ln2b[c]);
    }
    __syncthreads();

    // ---- g = gelu(hs @ fc1 + b) -> qs (f16, ld = C)
    for (int t = wv; t < 4 * (C / 16); t += 8) {
        int mt = t & 3, nt = t >> 2;
        v8f acc = {};
        for (int kk = 0; kk < C; kk += 32) {
            v16h a = frag_a_h(hs, C, mt * 16, kk, C, lane);
            v16h b = frag_b_gh(fc1W, C, kk, nt * 16, C, lane);
            acc = wmma16(a, b, acc);
        }
        int col = nt * 16 + (lane & 15);
        float bia = fc1b[col];
#pragma unroll
        for (int r = 0; r < 8; ++r) {
            int row = mt * 16 + r + 8 * (lane >> 4);
            qs[row * C + col] = (_Float16)gelu_tanh(acc[r] + bia);
        }
    }
    __syncthreads();

    // ---- x += g @ fc2 + b
    for (int t = wv; t < 4 * (C / 16); t += 8) {
        int mt = t & 3, nt = t >> 2;
        v8f acc = {};
        for (int kk = 0; kk < C; kk += 32) {
            v16h a = frag_a_h(qs, C, mt * 16, kk, C, lane);
            v16h b = frag_b_gh(fc2W, C, kk, nt * 16, C, lane);
            acc = wmma16(a, b, acc);
        }
        int col = nt * 16 + (lane & 15);
        float bia = fc2b[col];
#pragma unroll
        for (int r = 0; r < 8; ++r) {
            int row = mt * 16 + r + 8 * (lane >> 4);
            xs[row * C + col] += acc[r] + bia;
        }
    }
    __syncthreads();

    for (int i = tid; i < PATCH * C; i += 256) x[base + i] = xs[i];
}

// ---------------------------------------------------------------------------
// pooling: y[j] = max(lin(x[2j]), lin(x[2j+1])); then LN + GELU in place
// ---------------------------------------------------------------------------
__global__ void pool_kernel(const float* __restrict__ x, const float* __restrict__ W,
                            const float* __restrict__ bias, float* __restrict__ y,
                            int Cin, int Cout, long long total) {
    long long idx = (long long)blockIdx.x * 256 + threadIdx.x;
    if (idx >= total) return;
    int co = (int)(idx % Cout);
    long long j = idx / Cout;
    const float* r0 = x + 2 * j * Cin;
    const float* r1 = r0 + Cin;
    float y0 = bias[co], y1 = y0;
    for (int i = 0; i < Cin; ++i) {
        float w = W[i * Cout + co];
        y0 += r0[i] * w;
        y1 += r1[i] * w;
    }
    y[idx] = fmaxf(y0, y1);
}

__global__ void lngelu_kernel(float* __restrict__ y, const float* __restrict__ g,
                              const float* __restrict__ b, int C, long long rows) {
    long long r = (long long)blockIdx.x * 256 + threadIdx.x;
    if (r >= rows) return;
    float* p = y + r * C;
    float mu = 0.f;
    for (int c = 0; c < C; ++c) mu += p[c];
    mu /= C;
    float var = 0.f;
    for (int c = 0; c < C; ++c) { float d = p[c] - mu; var += d * d; }
    float rs = rsqrtf(var / C + 1e-5f);
    for (int c = 0; c < C; ++c)
        p[c] = gelu_tanh((p[c] - mu) * rs * g[c] + b[c]);
}

// ---------------------------------------------------------------------------
// decoder merge: x_out[r] = up(x_in[r/2]) + skip(enc[r])
// ---------------------------------------------------------------------------
__global__ void decmerge_kernel(const float* __restrict__ xin, const float* __restrict__ upW,
                                const float* __restrict__ upb, const float* __restrict__ enc,
                                const float* __restrict__ skW, const float* __restrict__ skb,
                                float* __restrict__ xout, int Cp, int Ce, int Co,
                                long long total) {
    long long idx = (long long)blockIdx.x * 256 + threadIdx.x;
    if (idx >= total) return;
    int co = (int)(idx % Co);
    long long r = idx / Co;
    const float* xi = xin + (r >> 1) * Cp;
    float s = upb[co] + skb[co];
    for (int i = 0; i < Cp; ++i) s += xi[i] * upW[i * Co + co];
    const float* er = enc + r * Ce;
    for (int i = 0; i < Ce; ++i) s += er[i] * skW[i * Co + co];
    xout[idx] = s;
}

// ---------------------------------------------------------------------------
// heads
// ---------------------------------------------------------------------------
__global__ void zero_kernel(float* p, int n) {
    int i = blockIdx.x * 256 + threadIdx.x;
    if (i < n) p[i] = 0.f;
}

__global__ void colsum_kernel(const float* __restrict__ x, long long rows, int C,
                              float* __restrict__ acc) {
    int c = threadIdx.x;
    if (c >= C) return;
    long long per = (rows + gridDim.x - 1) / gridDim.x;
    long long r0 = (long long)blockIdx.x * per;
    long long r1 = r0 + per;
    if (r1 > rows) r1 = rows;
    float s = 0.f;
    for (long long r = r0; r < r1; ++r) s += x[r * C + c];
    atomicAdd(&acc[c], s);
}

__global__ void heads_kernel(const float* __restrict__ acc, float invRows,
                             const float* W1, const float* b1, const float* W2,
                             const float* b2, const float* W3, const float* b3,
                             const float* n8g, const float* n8b, const float* clfW,
                             const float* clfb, const float* regW, const float* regb,
                             const float* sinW, const float* sinb, const float* cosW,
                             const float* cosb, float* __restrict__ out) {
    __shared__ float fm[96], e1[64], e2[32], e3[8], ne[8];
    int t = threadIdx.x;
    if (t < 96) fm[t] = acc[t] * invRows;
    __syncthreads();
    if (t < 64) {
        float s = b1[t];
        for (int i = 0; i < 96; ++i) s += fm[i] * W1[i * 64 + t];
        e1[t] = fmaxf(s, 0.f);
    }
    __syncthreads();
    if (t < 32) {
        float s = b2[t];
        for (int i = 0; i < 64; ++i) s += e1[i] * W2[i * 32 + t];
        e2[t] = fmaxf(s, 0.f);
    }
    __syncthreads();
    if (t < 8) {
        float s = b3[t];
        for (int i = 0; i < 32; ++i) s += e2[i] * W3[i * 8 + t];
        e3[t] = s;
    }
    __syncthreads();
    if (t == 0) {
        float mu = 0.f;
        for (int i = 0; i < 8; ++i) mu += e3[i];
        mu /= 8.f;
        float var = 0.f;
        for (int i = 0; i < 8; ++i) { float d = e3[i] - mu; var += d * d; }
        float rs = rsqrtf(var / 8.f + 1e-5f);
        for (int i = 0; i < 8; ++i) ne[i] = (e3[i] - mu) * rs * n8g[i] + n8b[i];
    }
    __syncthreads();
    if (t < 8) {
        float s = clfb[t];
        for (int i = 0; i < 8; ++i) s += ne[i] * clfW[i * 8 + t];
        out[t] = s;
    }
    if (t < 6) {
        float s = regb[t];
        for (int i = 0; i < 8; ++i) s += ne[i] * regW[i * 6 + t];
        out[8 + t] = s;
    }
    if (t == 0) {
        float s = sinb[0], c = cosb[0];
        for (int i = 0; i < 8; ++i) { s += ne[i] * sinW[i]; c += ne[i] * cosW[i]; }
        out[14] = tanhf(s);
        out[15] = tanhf(c);
    }
}

// ---------------------------------------------------------------------------
// launcher
// ---------------------------------------------------------------------------
extern "C" void kernel_launch(void* const* d_in, const int* in_sizes, int n_in,
                              void* d_out, int out_size, void* d_ws, size_t ws_size,
                              hipStream_t stream) {
    (void)in_sizes; (void)n_in; (void)out_size; (void)ws_size;
    auto F = [&](int i) { return (const float*)d_in[i]; };
    const float* points = F(0);

    // workspace carve-up
    char* ws = (char*)d_ws;
    size_t off = 0;
    auto carve = [&](size_t bytes) {
        void* p = ws + off;
        off += (bytes + 255) & ~(size_t)255;
        return p;
    };
    unsigned long long* keys = (unsigned long long*)carve((size_t)BN * 8);
    float* enc0 = (float*)carve((size_t)BN * 32 * 4);              // 262144 x 32
    float* enc1 = (float*)carve((size_t)(BN / 2) * 64 * 4);        // 131072 x 64
    float* enc2 = (float*)carve((size_t)(BN / 4) * 96 * 4);        //  65536 x 96
    float* enc3 = (float*)carve((size_t)(BN / 8) * 96 * 4);        //  32768 x 96
    float* enc4 = (float*)carve((size_t)(BN / 16) * 128 * 4);      //  16384 x 128
    float* dec3 = (float*)carve((size_t)(BN / 8) * 32 * 4);
    float* dec2 = (float*)carve((size_t)(BN / 4) * 64 * 4);
    float* dec1 = (float*)carve((size_t)(BN / 2) * 64 * 4);
    float* dec0 = (float*)carve((size_t)BN * 96 * 4);
    float* acc  = (float*)carve(96 * 4);

    // param index map (flattened dict order)
    const int ENC_B  = 3;                 // enc_blocks[i]: 11 arrays each
    const int POOL_B = 3 + 11 * 5;        // 58, enc_pool[i]: 4 arrays each
    const int DEC_B  = POOL_B + 4 * 4;    // 74, dec[s]: up.W,up.b,sk.W,sk.b + block(11)
    const int FEAT_B = DEC_B + 15 * 4;    // 134

    // ---- pre-convert the 9 blocks' GEMM weights to f16 (one-time, idempotent)
    struct BlkW { const _Float16 *wqkv, *wo, *fc1, *fc2; };
    BlkW bw[9];
    const int blkC[9]  = {32, 64, 96, 96, 128, 32, 64, 64, 96};
    const int blkPB[9] = {ENC_B + 0,  ENC_B + 11, ENC_B + 22, ENC_B + 33, ENC_B + 44,
                          DEC_B + 15 * 3 + 4, DEC_B + 15 * 2 + 4,
                          DEC_B + 15 * 1 + 4, DEC_B + 15 * 0 + 4};
    for (int i = 0; i < 9; ++i) {
        int C = blkC[i], pb = blkPB[i];
        int n1 = C * 3 * C, n2 = C * C;
        _Float16* q  = (_Float16*)carve((size_t)n1 * 2);
        _Float16* o  = (_Float16*)carve((size_t)n2 * 2);
        _Float16* f1 = (_Float16*)carve((size_t)n2 * 2);
        _Float16* f2 = (_Float16*)carve((size_t)n2 * 2);
        f32_to_f16_kernel<<<(n1 + 255) / 256, 256, 0, stream>>>(F(pb + 2), q, n1);
        f32_to_f16_kernel<<<(n2 + 255) / 256, 256, 0, stream>>>(F(pb + 3), o, n2);
        f32_to_f16_kernel<<<(n2 + 255) / 256, 256, 0, stream>>>(F(pb + 7), f1, n2);
        f32_to_f16_kernel<<<(n2 + 255) / 256, 256, 0, stream>>>(F(pb + 9), f2, n2);
        bw[i].wqkv = q; bw[i].wo = o; bw[i].fc1 = f1; bw[i].fc2 = f2;
    }

    // ---- Morton sort (bitonic, per batch) + gather + embed
    morton_init<<<BN / 256, 256, 0, stream>>>(points, keys);
    for (int k = 2; k <= NPTS; k <<= 1)
        for (int j = k >> 1; j > 0; j >>= 1)
            bitonic_step<<<BN / 256, 256, 0, stream>>>(keys, j, k);
    embed_gather<<<(BN * 32) / 256, 256, 0, stream>>>(points, keys, F(1), F(2), enc0);

#define LAUNCH_BLOCK(Cc, Hh, xbuf, npatch, pb, wi)                                     \
    do {                                                                               \
        size_t lds = (size_t)PATCH * (Cc) * 4 + (size_t)PATCH * (Cc) * 2 +             \
                     (size_t)PATCH * 3 * (Cc) * 2 + (size_t)(Hh) * PATCH * PATCH * 4;  \
        block_kernel<Cc, Hh><<<(npatch), 256, lds, stream>>>(                          \
            xbuf, F((pb) + 0), F((pb) + 1), bw[wi].wqkv, bw[wi].wo, F((pb) + 4),       \
            F((pb) + 5), F((pb) + 6), bw[wi].fc1, F((pb) + 8), bw[wi].fc2,             \
            F((pb) + 10));                                                             \
    } while (0)

    const long long R0 = BN, R1 = BN / 2, R2 = BN / 4, R3 = BN / 8, R4 = BN / 16;

    // ---- encoder
    LAUNCH_BLOCK(32, 1, enc0, (int)(R0 / 64), ENC_B + 11 * 0, 0);
    pool_kernel<<<(unsigned)((R1 * 64 + 255) / 256), 256, 0, stream>>>(
        enc0, F(POOL_B + 0), F(POOL_B + 1), enc1, 32, 64, R1 * 64);
    lngelu_kernel<<<(unsigned)((R1 + 255) / 256), 256, 0, stream>>>(
        enc1, F(POOL_B + 2), F(POOL_B + 3), 64, R1);
    LAUNCH_BLOCK(64, 2, enc1, (int)(R1 / 64), ENC_B + 11 * 1, 1);

    pool_kernel<<<(unsigned)((R2 * 96 + 255) / 256), 256, 0, stream>>>(
        enc1, F(POOL_B + 4), F(POOL_B + 5), enc2, 64, 96, R2 * 96);
    lngelu_kernel<<<(unsigned)((R2 + 255) / 256), 256, 0, stream>>>(
        enc2, F(POOL_B + 6), F(POOL_B + 7), 96, R2);
    LAUNCH_BLOCK(96, 4, enc2, (int)(R2 / 64), ENC_B + 11 * 2, 2);

    pool_kernel<<<(unsigned)((R3 * 96 + 255) / 256), 256, 0, stream>>>(
        enc2, F(POOL_B + 8), F(POOL_B + 9), enc3, 96, 96, R3 * 96);
    lngelu_kernel<<<(unsigned)((R3 + 255) / 256), 256, 0, stream>>>(
        enc3, F(POOL_B + 10), F(POOL_B + 11), 96, R3);
    LAUNCH_BLOCK(96, 4, enc3, (int)(R3 / 64), ENC_B + 11 * 3, 3);

    pool_kernel<<<(unsigned)((R4 * 128 + 255) / 256), 256, 0, stream>>>(
        enc3, F(POOL_B + 12), F(POOL_B + 13), enc4, 96, 128, R4 * 128);
    lngelu_kernel<<<(unsigned)((R4 + 255) / 256), 256, 0, stream>>>(
        enc4, F(POOL_B + 14), F(POOL_B + 15), 128, R4);
    LAUNCH_BLOCK(128, 4, enc4, (int)(R4 / 64), ENC_B + 11 * 4, 4);

    // ---- decoder (s = 3, 2, 1, 0)
    {
        int pb = DEC_B + 15 * 3;
        decmerge_kernel<<<(unsigned)((R3 * 32 + 255) / 256), 256, 0, stream>>>(
            enc4, F(pb + 0), F(pb + 1), enc3, F(pb + 2), F(pb + 3), dec3, 128, 96, 32,
            R3 * 32);
        LAUNCH_BLOCK(32, 2, dec3, (int)(R3 / 64), pb + 4, 5);
    }
    {
        int pb = DEC_B + 15 * 2;
        decmerge_kernel<<<(unsigned)((R2 * 64 + 255) / 256), 256, 0, stream>>>(
            dec3, F(pb + 0), F(pb + 1), enc2, F(pb + 2), F(pb + 3), dec2, 32, 96, 64,
            R2 * 64);
        LAUNCH_BLOCK(64, 2, dec2, (int)(R2 / 64), pb + 4, 6);
    }
    {
        int pb = DEC_B + 15 * 1;
        decmerge_kernel<<<(unsigned)((R1 * 64 + 255) / 256), 256, 0, stream>>>(
            dec2, F(pb + 0), F(pb + 1), enc1, F(pb + 2), F(pb + 3), dec1, 64, 64, 64,
            R1 * 64);
        LAUNCH_BLOCK(64, 2, dec1, (int)(R1 / 64), pb + 4, 7);
    }
    {
        int pb = DEC_B + 15 * 0;
        decmerge_kernel<<<(unsigned)((R0 * 96 + 255) / 256), 256, 0, stream>>>(
            dec1, F(pb + 0), F(pb + 1), enc0, F(pb + 2), F(pb + 3), dec0, 64, 32, 96,
            R0 * 96);
        LAUNCH_BLOCK(96, 4, dec0, (int)(R0 / 64), pb + 4, 8);
    }

    // ---- heads
    zero_kernel<<<1, 256, 0, stream>>>(acc, 96);
    colsum_kernel<<<512, 128, 0, stream>>>(dec0, R0, 96, acc);
    heads_kernel<<<1, 128, 0, stream>>>(
        acc, 1.f / (float)R0, F(FEAT_B + 0), F(FEAT_B + 1), F(FEAT_B + 2),
        F(FEAT_B + 3), F(FEAT_B + 4), F(FEAT_B + 5), F(140), F(141), F(142), F(143),
        F(144), F(145), F(146), F(147), F(148), F(149), (float*)d_out);
#undef LAUNCH_BLOCK
}
